// PointNetSetAbstraction_59717225284065
// MI455X (gfx1250) — matrix operand order repaired and checked by reading, not compiled, not run on hardware
//
#include <hip/hip_runtime.h>
#include <hip/hip_bf16.h>

typedef __bf16 bf16_t;
typedef __attribute__((ext_vector_type(8)))  __bf16 v8bf;
typedef __attribute__((ext_vector_type(16))) __bf16 v16bf;
typedef __attribute__((ext_vector_type(8)))  float  v8f;

#define BB   16
#define NN   4096
#define CC   64
#define SS   1024
#define KK   32
#define EPSV 1e-5f

#define OUT_NP_OFF (BB*SS*3)   // new_xyz [B,S,3] first, then [B,128,S]

// ---------------- workspace layout (bytes) ----------------
// 0       : w0b  bf16 [64][96]   (12288)
// 12288   : w1b  bf16 [64][64]   (8192)
// 20480   : w2b  bf16 [128][64]  (16384)
// 36864   : ab   f32  [512]      (alpha0,beta0,alpha1,beta1,alpha2[128],beta2[128])
// 38912   : new_xyz f32 [B][S][3] (196608)
// 235520  : nn_idx  i32 [B][S][K] (2097152)
// total ~2.23 MB

// =====================================================================
// Prep: fold BN into alpha/beta, convert + pad weights to bf16
// =====================================================================
__global__ void prep_kernel(const float* __restrict__ w0, const float* __restrict__ cb0,
                            const float* __restrict__ g0, const float* __restrict__ b0,
                            const float* __restrict__ m0, const float* __restrict__ v0,
                            const float* __restrict__ w1, const float* __restrict__ cb1,
                            const float* __restrict__ g1, const float* __restrict__ b1,
                            const float* __restrict__ m1, const float* __restrict__ v1,
                            const float* __restrict__ w2, const float* __restrict__ cb2,
                            const float* __restrict__ g2, const float* __restrict__ b2,
                            const float* __restrict__ m2, const float* __restrict__ v2,
                            bf16_t* __restrict__ w0b, bf16_t* __restrict__ w1b,
                            bf16_t* __restrict__ w2b, float* __restrict__ ab)
{
    int t = blockIdx.x * blockDim.x + threadIdx.x;
    int stride = gridDim.x * blockDim.x;
    for (int i = t; i < 64 * 96; i += stride) {
        int o = i / 96, k = i % 96;
        w0b[i] = (bf16_t)(k < 67 ? w0[o * 67 + k] : 0.0f);
    }
    for (int i = t; i < 64 * 64; i += stride)  w1b[i] = (bf16_t)w1[i];
    for (int i = t; i < 128 * 64; i += stride) w2b[i] = (bf16_t)w2[i];
    if (blockIdx.x == 0) {
        int l = threadIdx.x;
        if (l < 64) {
            float s0 = g0[l] * rsqrtf(v0[l] + EPSV);
            ab[l]       = s0;
            ab[64 + l]  = (cb0[l] - m0[l]) * s0 + b0[l];
            float s1 = g1[l] * rsqrtf(v1[l] + EPSV);
            ab[128 + l] = s1;
            ab[192 + l] = (cb1[l] - m1[l]) * s1 + b1[l];
        }
        if (l < 128) {
            float s2 = g2[l] * rsqrtf(v2[l] + EPSV);
            ab[256 + l] = s2;
            ab[384 + l] = (cb2[l] - m2[l]) * s2 + b2[l];
        }
    }
}

// =====================================================================
// FPS: one block per batch, points in registers, block argmax per step
// =====================================================================
__global__ __launch_bounds__(1024)
void fps_kernel(const float* __restrict__ xyz, float* __restrict__ nxyz_ws,
                float* __restrict__ out)
{
    __shared__ float s_val[32];
    __shared__ int   s_idx[32];
    __shared__ float s_cent[3];
    __shared__ int   s_far;

    int b = blockIdx.x;
    int t = threadIdx.x;
    int lane = t & 31, wave = t >> 5;
    const float* xb = xyz + (size_t)b * NN * 3;

    float px[4], py[4], pz[4], dist[4];
#pragma unroll
    for (int i = 0; i < 4; i++) {
        int p = i * 1024 + t;
        px[i] = xb[p * 3 + 0];
        py[i] = xb[p * 3 + 1];
        pz[i] = xb[p * 3 + 2];
        dist[i] = 1e10f;
    }
    if (t == 0) { s_far = 0; s_cent[0] = xb[0]; s_cent[1] = xb[1]; s_cent[2] = xb[2]; }
    __syncthreads();

    for (int it = 0; it < SS; ++it) {
        float cx = s_cent[0], cy = s_cent[1], cz = s_cent[2];
        if (t == 0) {
            size_t o = ((size_t)b * SS + it) * 3;
            nxyz_ws[o + 0] = cx; nxyz_ws[o + 1] = cy; nxyz_ws[o + 2] = cz;
            out[o + 0] = cx;     out[o + 1] = cy;     out[o + 2] = cz;
        }
        float bestv = -1.0f; int besti = 0;
#pragma unroll
        for (int i = 0; i < 4; i++) {
            float dx = px[i] - cx, dy = py[i] - cy, dz = pz[i] - cz;
            float d = dx * dx + dy * dy + dz * dz;
            dist[i] = fminf(dist[i], d);
            int idx = i * 1024 + t;
            if (dist[i] > bestv || (dist[i] == bestv && idx < besti)) { bestv = dist[i]; besti = idx; }
        }
#pragma unroll
        for (int off = 16; off >= 1; off >>= 1) {
            float ov = __shfl_xor(bestv, off, 32);
            int   oi = __shfl_xor(besti, off, 32);
            if (ov > bestv || (ov == bestv && oi < besti)) { bestv = ov; besti = oi; }
        }
        if (lane == 0) { s_val[wave] = bestv; s_idx[wave] = besti; }
        __syncthreads();
        if (wave == 0) {
            bestv = s_val[lane]; besti = s_idx[lane];
#pragma unroll
            for (int off = 16; off >= 1; off >>= 1) {
                float ov = __shfl_xor(bestv, off, 32);
                int   oi = __shfl_xor(besti, off, 32);
                if (ov > bestv || (ov == bestv && oi < besti)) { bestv = ov; besti = oi; }
            }
            if (lane == 0) s_far = besti;
        }
        __syncthreads();
        int nf = s_far;
        if (t == (nf & 1023)) {
            int i = nf >> 10;
            float sx = px[0], sy = py[0], sz = pz[0];
            if (i == 1) { sx = px[1]; sy = py[1]; sz = pz[1]; }
            if (i == 2) { sx = px[2]; sy = py[2]; sz = pz[2]; }
            if (i == 3) { sx = px[3]; sy = py[3]; sz = pz[3]; }
            s_cent[0] = sx; s_cent[1] = sy; s_cent[2] = sz;
        }
        __syncthreads();
    }
}

// =====================================================================
// KNN: 1 thread per query, candidate tile in LDS, sorted K-list in LDS
// =====================================================================
#define KNT 128
#define KTILE 1024
__global__ __launch_bounds__(KNT)
void knn_kernel(const float* __restrict__ xyz, const float* __restrict__ nxyz,
                int* __restrict__ nn_idx)
{
    __shared__ float s_x[KTILE], s_y[KTILE], s_z[KTILE];
    __shared__ float s_d[KK * KNT];
    __shared__ int   s_i[KK * KNT];

    int b = blockIdx.y;
    int t = threadIdx.x;
    int s = blockIdx.x * KNT + t;
    const float* q = nxyz + ((size_t)b * SS + s) * 3;
    float qx = q[0], qy = q[1], qz = q[2];
#pragma unroll
    for (int k = 0; k < KK; k++) s_d[k * KNT + t] = 3.4e38f;
    float worst = 3.4e38f;

    const float* xb = xyz + (size_t)b * NN * 3;
    for (int base = 0; base < NN; base += KTILE) {
        __syncthreads();
        for (int i = t; i < KTILE; i += KNT) {
            const float* p = xb + (size_t)(base + i) * 3;
            s_x[i] = p[0]; s_y[i] = p[1]; s_z[i] = p[2];
        }
        __syncthreads();
        for (int j = 0; j < KTILE; j++) {
            float dx = s_x[j] - qx, dy = s_y[j] - qy, dz = s_z[j] - qz;
            float d = dx * dx + dy * dy + dz * dz;
            if (d < worst) {
                int idx = base + j;
                int k = KK - 1;
                while (k > 0 && s_d[(k - 1) * KNT + t] > d) {
                    s_d[k * KNT + t] = s_d[(k - 1) * KNT + t];
                    s_i[k * KNT + t] = s_i[(k - 1) * KNT + t];
                    k--;
                }
                s_d[k * KNT + t] = d;
                s_i[k * KNT + t] = idx;
                worst = s_d[(KK - 1) * KNT + t];
            }
        }
    }
    size_t o = ((size_t)b * SS + s) * KK;
#pragma unroll
    for (int k = 0; k < KK; k++) nn_idx[o + k] = s_i[k * KNT + t];
}

// =====================================================================
// Fused gather + 3x (GEMM, BN, ReLU) via WMMA bf16 + max-over-K
// =====================================================================
#define TQ 4
#define ROWS 128        // TQ * K
#define LDA0 96
#define LDA1 64

union V16U { v16bf v; v8bf h[2]; };
static __device__ inline v16bf mk16(v8bf lo, v8bf hi) { V16U u; u.h[0] = lo; u.h[1] = hi; return u.v; }
// 16-bit A-matrix 16x32 fragment: lane half 'hi' selects K run {16h..16h+15}
static __device__ inline v16bf load_afrag(const bf16_t* p, int hi) {
    const bf16_t* q = p + 16 * hi;
    return mk16(*(const v8bf*)q, *(const v8bf*)(q + 8));
}

__global__ __launch_bounds__(256)
void mlp_kernel(const float* __restrict__ xyz, const float* __restrict__ points,
                const float* __restrict__ nxyz, const int* __restrict__ nn_idx,
                const bf16_t* __restrict__ w0b, const bf16_t* __restrict__ w1b,
                const bf16_t* __restrict__ w2b, const float* __restrict__ ab,
                float* __restrict__ out)
{
    __shared__ __align__(16) bf16_t A0[ROWS * LDA0];
    __shared__ __align__(16) bf16_t A1[ROWS * LDA1];
    __shared__ float part[8][128];

    int b = blockIdx.y;
    int sq0 = blockIdx.x * TQ;
    int t = threadIdx.x;

    // ---- gather: one thread per row (q,k) -> bf16 features in LDS ----
    if (t < ROWS) {
        int q = t >> 5, k = t & 31;
        int s = sq0 + q;
        int idx = nn_idx[((size_t)b * SS + s) * KK + k];
        const float* p  = xyz  + ((size_t)b * NN + idx) * 3;
        const float* nq = nxyz + ((size_t)b * SS + s) * 3;
        bf16_t* row = &A0[t * LDA0];
        row[0] = (bf16_t)(p[0] - nq[0]);
        row[1] = (bf16_t)(p[1] - nq[1]);
        row[2] = (bf16_t)(p[2] - nq[2]);
        const float4* pr = (const float4*)(points + ((size_t)b * NN + idx) * CC);
#pragma unroll
        for (int i = 0; i < 16; i++) {
            float4 f = pr[i];
            row[3 + 4 * i + 0] = (bf16_t)f.x;
            row[3 + 4 * i + 1] = (bf16_t)f.y;
            row[3 + 4 * i + 2] = (bf16_t)f.z;
            row[3 + 4 * i + 3] = (bf16_t)f.w;
        }
#pragma unroll
        for (int i = 67; i < 96; i++) row[i] = (bf16_t)0.0f;
    }
    __syncthreads();

    int wave = t >> 5, lane = t & 31;
    int hi = lane >> 4;      // lane half
    int ln = lane & 15;      // row (A) / column (B,C,D) within tile
    int row0 = wave * 16;    // this wave's 16-row strip

    const float* alpha0 = ab;       const float* beta0 = ab + 64;
    const float* alpha1 = ab + 128; const float* beta1 = ab + 192;
    const float* alpha2 = ab + 256; const float* beta2 = ab + 384;

    // ---------------- layer 0: 67(->96) -> 64 ----------------
    v16bf a0[3];
#pragma unroll
    for (int c = 0; c < 3; c++)
        a0[c] = load_afrag(&A0[(row0 + ln) * LDA0 + c * 32], hi);

#pragma unroll
    for (int nt = 0; nt < 4; nt++) {
        v8f acc = {};
#pragma unroll
        for (int c = 0; c < 3; c++) {
            const bf16_t* wr = w0b + (size_t)(nt * 16 + ln) * 96 + c * 32 + 16 * hi;
            v16bf bfrag = *(const v16bf*)wr;
            acc = __builtin_amdgcn_wmma_f32_16x16x32_bf16(false, a0[c], false, bfrag,
                                                          (short)0, acc, false, false);
        }
        float al = alpha0[nt * 16 + ln], be = beta0[nt * 16 + ln];
#pragma unroll
        for (int v = 0; v < 8; v++) {
            float y = fmaxf(acc[v] * al + be, 0.0f);
            A1[(row0 + v + 8 * hi) * LDA1 + nt * 16 + ln] = (bf16_t)y;
        }
    }

    // ---------------- layer 1: 64 -> 64 (in-place in A1) ----------------
    v16bf a1[2];
#pragma unroll
    for (int c = 0; c < 2; c++)
        a1[c] = load_afrag(&A1[(row0 + ln) * LDA1 + c * 32], hi);

#pragma unroll
    for (int nt = 0; nt < 4; nt++) {
        v8f acc = {};
#pragma unroll
        for (int c = 0; c < 2; c++) {
            const bf16_t* wr = w1b + (size_t)(nt * 16 + ln) * 64 + c * 32 + 16 * hi;
            v16bf bfrag = *(const v16bf*)wr;
            acc = __builtin_amdgcn_wmma_f32_16x16x32_bf16(false, a1[c], false, bfrag,
                                                          (short)0, acc, false, false);
        }
        float al = alpha1[nt * 16 + ln], be = beta1[nt * 16 + ln];
#pragma unroll
        for (int v = 0; v < 8; v++) {
            float y = fmaxf(acc[v] * al + be, 0.0f);
            A1[(row0 + v + 8 * hi) * LDA1 + nt * 16 + ln] = (bf16_t)y;
        }
    }

    // ---------------- layer 2: 64 -> 128 + max over rows ----------------
    v16bf a2[2];
#pragma unroll
    for (int c = 0; c < 2; c++)
        a2[c] = load_afrag(&A1[(row0 + ln) * LDA1 + c * 32], hi);

#pragma unroll
    for (int nt = 0; nt < 8; nt++) {
        v8f acc = {};
#pragma unroll
        for (int c = 0; c < 2; c++) {
            const bf16_t* wr = w2b + (size_t)(nt * 16 + ln) * 64 + c * 32 + 16 * hi;
            v16bf bfrag = *(const v16bf*)wr;
            acc = __builtin_amdgcn_wmma_f32_16x16x32_bf16(false, a2[c], false, bfrag,
                                                          (short)0, acc, false, false);
        }
        float al = alpha2[nt * 16 + ln], be = beta2[nt * 16 + ln];
        float m = fmaxf(acc[0] * al + be, 0.0f);
#pragma unroll
        for (int v = 1; v < 8; v++) m = fmaxf(m, fmaxf(acc[v] * al + be, 0.0f));
        m = fmaxf(m, __shfl_xor(m, 16, 32));          // combine lane halves -> 16 rows
        if (lane < 16) part[wave][nt * 16 + ln] = m;
    }
    __syncthreads();

    // combine the two 16-row halves of each query, write [B,128,S]
    for (int u = t; u < TQ * 128; u += 256) {
        int q = u >> 7, c = u & 127;
        float m = fmaxf(part[2 * q][c], part[2 * q + 1][c]);
        out[OUT_NP_OFF + ((size_t)b * 128 + c) * SS + (sq0 + q)] = m;
    }
}

// =====================================================================
extern "C" void kernel_launch(void* const* d_in, const int* in_sizes, int n_in,
                              void* d_out, int out_size, void* d_ws, size_t ws_size,
                              hipStream_t stream)
{
    const float* xyz    = (const float*)d_in[0];
    const float* points = (const float*)d_in[1];
    const float* w0 = (const float*)d_in[2];  const float* cb0 = (const float*)d_in[3];
    const float* g0 = (const float*)d_in[4];  const float* b0  = (const float*)d_in[5];
    const float* m0 = (const float*)d_in[6];  const float* v0  = (const float*)d_in[7];
    const float* w1 = (const float*)d_in[8];  const float* cb1 = (const float*)d_in[9];
    const float* g1 = (const float*)d_in[10]; const float* b1  = (const float*)d_in[11];
    const float* m1 = (const float*)d_in[12]; const float* v1  = (const float*)d_in[13];
    const float* w2 = (const float*)d_in[14]; const float* cb2 = (const float*)d_in[15];
    const float* g2 = (const float*)d_in[16]; const float* b2  = (const float*)d_in[17];
    const float* m2 = (const float*)d_in[18]; const float* v2  = (const float*)d_in[19];

    float* out = (float*)d_out;
    char* ws = (char*)d_ws;
    bf16_t* w0b = (bf16_t*)(ws + 0);
    bf16_t* w1b = (bf16_t*)(ws + 12288);
    bf16_t* w2b = (bf16_t*)(ws + 20480);
    float*  ab  = (float*)(ws + 36864);
    float*  nx  = (float*)(ws + 38912);
    int*    nni = (int*)(ws + 235520);

    prep_kernel<<<dim3(32), dim3(256), 0, stream>>>(
        w0, cb0, g0, b0, m0, v0, w1, cb1, g1, b1, m1, v1,
        w2, cb2, g2, b2, m2, v2, w0b, w1b, w2b, ab);

    fps_kernel<<<dim3(BB), dim3(1024), 0, stream>>>(xyz, nx, out);

    knn_kernel<<<dim3(SS / KNT, BB), dim3(KNT), 0, stream>>>(xyz, nx, nni);

    mlp_kernel<<<dim3(SS / TQ, BB), dim3(256), 0, stream>>>(
        xyz, points, nx, nni, w0b, w1b, w2b, ab, out);
}
